// BiAttnConv_29824252903955
// MI455X (gfx1250) — matrix operand order repaired and checked by reading, not compiled, not run on hardware
//
#include <hip/hip_runtime.h>

typedef __attribute__((ext_vector_type(16))) _Float16 v16h;
typedef __attribute__((ext_vector_type(8)))  _Float16 v8h;
typedef __attribute__((ext_vector_type(8)))  float    v8f;
typedef __attribute__((__vector_size__(4 * sizeof(int)))) int v4i;

#define GEMM_RELU 1
#define GEMM_BETA 2

#if defined(__has_builtin)
#if __has_builtin(__builtin_amdgcn_global_load_async_to_lds_b128)
#define HAVE_ASYNC_LDS 1
#endif
#endif
#ifndef HAVE_ASYNC_LDS
#define HAVE_ASYNC_LDS 0
#endif

#if defined(__has_builtin)
#if __has_builtin(__builtin_amdgcn_s_wait_asynccnt)
#define WAIT_ASYNC() __builtin_amdgcn_s_wait_asynccnt(0)
#else
#define WAIT_ASYNC() asm volatile("s_wait_asynccnt 0" ::: "memory")
#endif
#else
#define WAIT_ASYNC() asm volatile("s_wait_asynccnt 0" ::: "memory")
#endif

// Address-space helpers typed for the async-to-LDS builtin (param type from the
// compiler diagnostic: v4i in __device__/global AS, LDS side in __shared__ AS).
// LDS generic pointers carry the 32-bit LDS offset in the low bits, so
// truncation through uintptr yields a valid addrspace(3) pointer.
__device__ __forceinline__ __attribute__((address_space(1))) v4i* as_global_v4(const void* p)
{
    return (__attribute__((address_space(1))) v4i*)(unsigned long long)(p);
}
__device__ __forceinline__ __attribute__((address_space(3))) v4i* as_lds_v4(void* p)
{
    return (__attribute__((address_space(3))) v4i*)(unsigned)(unsigned long long)(p);
}

// C[M,N] = act( A[M,K](lda,f16) * W[N,K](ldw,f16)^T + bias [+ C] )
// Block tile 64x64, 8 waves: wave grid 4(M) x 2(N), each wave 16x32 (2 WMMAs
// sharing one A fragment). K chunked by 32, double-buffered LDS; tiles arrive
// via async global->LDS DMA (ASYNCcnt) when available.
__global__ __launch_bounds__(256) void gemm_wmma_kernel(
    const _Float16* __restrict__ A, int lda,
    const _Float16* __restrict__ W, int ldw,
    const float* __restrict__ bias,
    float* __restrict__ C, int ldc,
    _Float16* __restrict__ Ch,           // if non-null, write f16 output instead
    int M, int N, int K, int flags)
{
    __shared__ __align__(16) _Float16 As[2][64 * 40];   // 80B row stride, 16B slots
    __shared__ __align__(16) _Float16 Bs[2][64 * 40];

    const int tid  = threadIdx.x;
    const int bm   = blockIdx.y * 64;
    const int bn   = blockIdx.x * 64;
    const int wave = tid >> 5;
    const int lane = tid & 31;
    const int wm   = (wave >> 1) << 4;   // 0,16,32,48
    const int wn   = (wave & 1)  << 5;   // 0,32
    const int half = lane >> 4;
    const int l15  = lane & 15;

    const int lrow = tid >> 2;           // 0..63
    const int lcol = (tid & 3) << 3;     // 0,8,16,24 (f16 elems -> 16B chunks)

    const int ar = (bm + lrow < M) ? (bm + lrow) : (M - 1);
    const int br = (bn + lrow < N) ? (bn + lrow) : (N - 1);
    const _Float16* agp = A + (size_t)ar * lda + lcol;
    const _Float16* bgp = W + (size_t)br * ldw + lcol;
    const int lofs = lrow * 40 + lcol;   // f16 units; bytes = 80*row + 16*chunk

    v8f acc0 = {}, acc1 = {};

    auto issue = [&](int p, int k0) {
#if HAVE_ASYNC_LDS
        __builtin_amdgcn_global_load_async_to_lds_b128(
            as_global_v4(agp + k0), as_lds_v4(&As[p][lofs]), 0, 0);
        __builtin_amdgcn_global_load_async_to_lds_b128(
            as_global_v4(bgp + k0), as_lds_v4(&Bs[p][lofs]), 0, 0);
#else
        *(v8h*)&As[p][lofs] = *(const v8h*)(agp + k0);
        *(v8h*)&Bs[p][lofs] = *(const v8h*)(bgp + k0);
#endif
    };

    issue(0, 0);
    int p = 0;
    for (int k0 = 0; k0 < K; k0 += 32) {
#if HAVE_ASYNC_LDS
        WAIT_ASYNC();
#endif
        __syncthreads();

        // A frag: row = wm + l15; K = 16*half + e
        v8h a0  = *(const v8h*)&As[p][(wm + l15) * 40 + (half << 4)];
        v8h a1  = *(const v8h*)&As[p][(wm + l15) * 40 + (half << 4) + 8];
        // B frags: col = wn (+16) + l15; K = 16*half + e
        v8h b00 = *(const v8h*)&Bs[p][(wn + l15) * 40 + (half << 4)];
        v8h b01 = *(const v8h*)&Bs[p][(wn + l15) * 40 + (half << 4) + 8];
        v8h b10 = *(const v8h*)&Bs[p][(wn + 16 + l15) * 40 + (half << 4)];
        v8h b11 = *(const v8h*)&Bs[p][(wn + 16 + l15) * 40 + (half << 4) + 8];

        if (k0 + 32 < K) issue(p ^ 1, k0 + 32);   // overlap DMA with WMMA

        v16h av, bv0, bv1;
#pragma unroll
        for (int i = 0; i < 8; ++i) {
            av[i]  = a0[i];  av[i + 8]  = a1[i];
            bv0[i] = b00[i]; bv0[i + 8] = b01[i];
            bv1[i] = b10[i]; bv1[i + 8] = b11[i];
        }
        acc0 = __builtin_amdgcn_wmma_f32_16x16x32_f16(false, av, false, bv0,
                                                      (short)0, acc0, false, false);
        acc1 = __builtin_amdgcn_wmma_f32_16x16x32_f16(false, av, false, bv1,
                                                      (short)0, acc1, false, false);
        p ^= 1;
    }

#pragma unroll
    for (int t = 0; t < 2; ++t) {
        v8f acc = t ? acc1 : acc0;
        const int cn = bn + wn + (t << 4) + l15;
        if (cn < N) {
            const float bb = bias ? bias[cn] : 0.0f;
#pragma unroll
            for (int r = 0; r < 8; ++r) {
                const int cm = bm + wm + (half << 3) + r;
                if (cm < M) {
                    float v = acc[r] + bb;
                    if (flags & GEMM_BETA) v += C[(size_t)cm * ldc + cn];
                    if (flags & GEMM_RELU) v = v > 0.0f ? v : 0.0f;
                    if (Ch) Ch[(size_t)cm * ldc + cn] = (_Float16)v;
                    else    C [(size_t)cm * ldc + cn] = v;
                }
            }
        }
    }
}

// 3x3 local attention; one thread per (b, y, x, head); hd = 32, nh = 8.
// Output written as f16 (sole consumer is the WMMA out-projection GEMM).
__global__ void local_attn_kernel(
    const float* __restrict__ qsrc, int qoff, int qP, int qHs, int qWs, int qstep,
    const float* __restrict__ kv, int koff, int voff, int kvP, int Hs, int Ws, int stride,
    const float* __restrict__ pos,   // [256, 9] slice base
    _Float16* __restrict__ out,      // [B, Ho, Wo, 256]
    int Bn, int Ho, int Wo, float scale)
{
    int idx = blockIdx.x * blockDim.x + threadIdx.x;
    int total = Bn * Ho * Wo * 8;
    if (idx >= total) return;
    int h = idx & 7;
    int t = idx >> 3;
    int x = t % Wo; t /= Wo;
    int y = t % Ho;
    int b = t / Ho;
    int c0 = h << 5;

    const float* qp = qsrc + ((size_t)(b * qHs + y * qstep) * qWs + x * qstep) * qP + qoff + c0;
    float qv[32];
#pragma unroll
    for (int d = 0; d < 32; ++d) qv[d] = qp[d] * scale;

    float lg[9];
    bool inb[9];
    for (int w = 0; w < 9; ++w) {
        int dy = w / 3 - 1, dx = w % 3 - 1;
        int sy = y * stride + dy, sx = x * stride + dx;
        bool in = (sy >= 0) & (sy < Hs) & (sx >= 0) & (sx < Ws);
        inb[w] = in;
        int cy = in ? sy : 0, cx = in ? sx : 0;   // clamp -> always-valid address
        const float* kp = kv + ((size_t)(b * Hs + cy) * Ws + cx) * kvP + koff + c0;
        const float* pp = pos + (size_t)c0 * 9 + w;
        float a = 0.0f;
#pragma unroll
        for (int d = 0; d < 32; ++d) {
            float kd = in ? kp[d] : 0.0f;   // zero-pad window, pos bias still applies
            a += qv[d] * (kd + pp[(size_t)d * 9]);
        }
        lg[w] = a;
    }
    float mx = lg[0];
#pragma unroll
    for (int w = 1; w < 9; ++w) mx = fmaxf(mx, lg[w]);
    float s = 0.0f;
#pragma unroll
    for (int w = 0; w < 9; ++w) { lg[w] = __expf(lg[w] - mx); s += lg[w]; }
    float inv = 1.0f / s;

    float o[32];
#pragma unroll
    for (int d = 0; d < 32; ++d) o[d] = 0.0f;
    for (int w = 0; w < 9; ++w) {
        float aw = lg[w] * inv;
        int dy = w / 3 - 1, dx = w % 3 - 1;
        int sy = y * stride + dy, sx = x * stride + dx;
        bool in = inb[w];
        int cy = in ? sy : 0, cx = in ? sx : 0;
        const float* vp = kv + ((size_t)(b * Hs + cy) * Ws + cx) * kvP + voff + c0;
#pragma unroll
        for (int d = 0; d < 32; ++d) {
            float vd = in ? vp[d] : 0.0f;
            o[d] += aw * vd;
        }
    }
    _Float16* op = out + ((size_t)(b * Ho + y) * Wo + x) * 256 + c0;
#pragma unroll
    for (int d = 0; d < 32; ++d) op[d] = (_Float16)o[d];
}

__global__ void copy_kernel(const float* __restrict__ src, float* __restrict__ dst, int n)
{
    int i = blockIdx.x * blockDim.x + threadIdx.x;
    if (i < n) dst[i] = src[i];
}

__global__ void f32_to_f16_kernel(const float* __restrict__ src,
                                  _Float16* __restrict__ dst, int n)
{
    int i = blockIdx.x * blockDim.x + threadIdx.x;
    if (i < n) dst[i] = (_Float16)src[i];
}

// acc[B, 2Hc, 2Wc, 256] += bilinear-align-corners 2x upsample of src padded by one
// zero row/col (matches pad + _up2_ac + crop in the reference).
__global__ void up_add_kernel(const float* __restrict__ src, float* __restrict__ acc,
                              int Bn, int Hc, int Wc)
{
    int Hf = Hc * 2, Wf = Wc * 2;
    int idx = blockIdx.x * blockDim.x + threadIdx.x;
    int total = Bn * Hf * Wf * 256;
    if (idx >= total) return;
    int c = idx & 255;
    int t = idx >> 8;
    int x = t % Wf; t /= Wf;
    int y = t % Hf;
    int b = t / Hf;
    int y0 = y >> 1, x0 = x >> 1;
    size_t base = ((size_t)b * Hc * Wc) * 256 + c;
    auto val = [&](int cy, int cx) -> float {
        return (cy < Hc && cx < Wc) ? src[base + ((size_t)cy * Wc + cx) * 256] : 0.0f;
    };
    float v;
    if (!(y & 1) && !(x & 1)) v = val(y0, x0);
    else if (!(y & 1))        v = 0.5f * (val(y0, x0) + val(y0, x0 + 1));
    else if (!(x & 1))        v = 0.5f * (val(y0, x0) + val(y0 + 1, x0));
    else                      v = 0.25f * (val(y0, x0) + val(y0, x0 + 1) +
                                           val(y0 + 1, x0) + val(y0 + 1, x0 + 1));
    acc[idx] += v;
}

// LayerNorm over 256 channels; one wave32 per row; optional second addend x2;
// optional extra f16 copy of the output (for a downstream WMMA GEMM).
__global__ void ln_kernel(const float* __restrict__ x, const float* __restrict__ x2,
                          const float* __restrict__ g, const float* __restrict__ be,
                          float* __restrict__ out, _Float16* __restrict__ outh, int M)
{
    int gw = (blockIdx.x * blockDim.x + threadIdx.x) >> 5;
    int lane = threadIdx.x & 31;
    if (gw >= M) return;
    size_t base = (size_t)gw * 256;
    float v[8];
    float s = 0.0f;
#pragma unroll
    for (int i = 0; i < 8; ++i) {
        int c = lane + (i << 5);
        float t = x[base + c];
        if (x2) t += x2[base + c];
        v[i] = t; s += t;
    }
#pragma unroll
    for (int o = 16; o > 0; o >>= 1) s += __shfl_xor(s, o, 32);
    float mu = s * (1.0f / 256.0f);
    float var = 0.0f;
#pragma unroll
    for (int i = 0; i < 8; ++i) { float d = v[i] - mu; var += d * d; }
#pragma unroll
    for (int o = 16; o > 0; o >>= 1) var += __shfl_xor(var, o, 32);
    var *= (1.0f / 256.0f);
    float inv = rsqrtf(var + 1e-5f);
#pragma unroll
    for (int i = 0; i < 8; ++i) {
        int c = lane + (i << 5);
        float r = (v[i] - mu) * inv * g[c] + be[c];
        out[base + c] = r;
        if (outh) outh[base + c] = (_Float16)r;
    }
}

extern "C" void kernel_launch(void* const* d_in, const int* in_sizes, int n_in,
                              void* d_out, int out_size, void* d_ws, size_t ws_size,
                              hipStream_t stream)
{
    (void)in_sizes; (void)n_in; (void)out_size; (void)ws_size;
    const float* fm[3]    = {(const float*)d_in[0],  (const float*)d_in[1],  (const float*)d_in[2]};
    const float* in_w[3]  = {(const float*)d_in[3],  (const float*)d_in[4],  (const float*)d_in[5]};
    const float* in_b[3]  = {(const float*)d_in[6],  (const float*)d_in[7],  (const float*)d_in[8]};
    const float* pos[3]   = {(const float*)d_in[9],  (const float*)d_in[10], (const float*)d_in[11]};
    const float* out_w[3] = {(const float*)d_in[12], (const float*)d_in[13], (const float*)d_in[14]};
    const float* out_b[3] = {(const float*)d_in[15], (const float*)d_in[16], (const float*)d_in[17]};
    const float* ln1_w[3] = {(const float*)d_in[18], (const float*)d_in[19], (const float*)d_in[20]};
    const float* ln1_b[3] = {(const float*)d_in[21], (const float*)d_in[22], (const float*)d_in[23]};
    const float* f1_w[3]  = {(const float*)d_in[24], (const float*)d_in[25], (const float*)d_in[26]};
    const float* f1_b[3]  = {(const float*)d_in[27], (const float*)d_in[28], (const float*)d_in[29]};
    const float* f2_w[3]  = {(const float*)d_in[30], (const float*)d_in[31], (const float*)d_in[32]};
    const float* f2_b[3]  = {(const float*)d_in[33], (const float*)d_in[34], (const float*)d_in[35]};
    const float* ln2_w[3] = {(const float*)d_in[36], (const float*)d_in[37], (const float*)d_in[38]};
    const float* ln2_b[3] = {(const float*)d_in[39], (const float*)d_in[40], (const float*)d_in[41]};

    const int B = 2;
    const int Hd[3]    = {64, 32, 16};
    const int Mrows[3] = {8192, 2048, 512};
    const int Pw[3]    = {1536, 2304, 1536};
    const float scale  = 0.17677669529663687f;  // 32^-0.5

    float* ws = (float*)d_ws;
    // f32 regions (units: floats)
    float* P[3]   = { ws,               ws + 12582912ull, ws + 17301504ull };
    float* ACC[3] = { ws + 18087936ull, ws + 20185088ull, ws + 20709376ull };
    float* T2     = ws + 20840448ull;                 // 2,097,152 floats
    float* XBUF   = ws + 12582912ull;                 // aliases P1 (FFN phase only)
    // f16 regions
    _Float16* T1h = (_Float16*)(ws + 22937600ull);    // 2,097,152 f16
    _Float16* FMH = (_Float16*)(ws + 23986176ull);    // 2,752,512 f16
    _Float16* WH  = (_Float16*)(ws + 25362432ull);    // 3,538,944 f16
    _Float16* XBh = (_Float16*)(ws + 14680064ull);    // aliases P1 tail (FFN only)
    _Float16* HBh = (_Float16*)(ws + 0ull);           // aliases P0 (FFN only)

    _Float16* FMh[3]  = { FMH, FMH + 2097152ull, FMH + 2621440ull };
    _Float16* inWh[3] = { WH, WH + 393216ull, WH + 983040ull };
    _Float16* outWh[3], *f1Wh[3], *f2Wh[3];
    for (int i = 0; i < 3; ++i) {
        outWh[i] = WH + 1376256ull + (size_t)i * 196608ull;
        f1Wh[i]  = WH + 1966080ull + (size_t)i * 262144ull;
        f2Wh[i]  = WH + 2752512ull + (size_t)i * 262144ull;
    }
    float* outp = (float*)d_out;
    size_t outoff[3] = {0ull, 2097152ull, 2621440ull};

    auto cvt = [&](const float* s, _Float16* d, int n) {
        f32_to_f16_kernel<<<(n + 255) / 256, 256, 0, stream>>>(s, d, n);
    };
    auto gemm = [&](const _Float16* A, int lda, const _Float16* W, int ldw,
                    const float* bias, float* C, _Float16* Ch, int ldc,
                    int M, int N, int K, int flags) {
        dim3 grid((N + 63) / 64, (M + 63) / 64);
        gemm_wmma_kernel<<<grid, 256, 0, stream>>>(A, lda, W, ldw, bias, C, ldc, Ch,
                                                   M, N, K, flags);
    };

    // 0) one-time f16 conversions (weights + residual inputs)
    for (int i = 0; i < 3; ++i) cvt(fm[i],    FMh[i],  Mrows[i] * 256);
    for (int i = 0; i < 3; ++i) cvt(in_w[i],  inWh[i], Pw[i] * 256);
    for (int i = 0; i < 3; ++i) cvt(out_w[i], outWh[i], 196608);
    for (int i = 0; i < 3; ++i) cvt(f1_w[i],  f1Wh[i], 262144);
    for (int i = 0; i < 3; ++i) cvt(f2_w[i],  f2Wh[i], 262144);

    // 1) input projections (f32 output; consumed by VALU attention)
    for (int i = 0; i < 3; ++i)
        gemm(FMh[i], 256, inWh[i], 256, in_b[i], P[i], nullptr, Pw[i],
             Mrows[i], Pw[i], 256, 0);

    // 2) accumulator = residual input
    for (int i = 0; i < 3; ++i) {
        int n = Mrows[i] * 256;
        copy_kernel<<<(n + 255) / 256, 256, 0, stream>>>(fm[i], ACC[i], n);
    }

    // 3) self attention (stride-1, same map)
    for (int i = 0; i < 3; ++i) {
        int H = Hd[i], tot = B * H * H * 8;
        local_attn_kernel<<<(tot + 255) / 256, 256, 0, stream>>>(
            P[i], 0, Pw[i], H, H, 1,
            P[i], 256, 512, Pw[i], H, H, 1,
            pos[i], T1h, B, H, H, scale);
        gemm(T1h, 256, outWh[i], 768, out_b[i], ACC[i], nullptr, 256,
             Mrows[i], 256, 256, GEMM_BETA);
    }

    // 4) top-down: strided fine queries vs coarse K/V, 2x align-corners upsample add
    for (int i = 0; i < 2; ++i) {
        int H = Hd[i], Hc = Hd[i + 1];
        int koff = (i + 1 == 2) ? 768 : 1024;
        int tot = B * Hc * Hc * 8;
        local_attn_kernel<<<(tot + 255) / 256, 256, 0, stream>>>(
            P[i], 768, Pw[i], H, H, 2,
            P[i + 1], koff, koff + 256, Pw[i + 1], Hc, Hc, 1,
            pos[i] + 256 * 9, T1h, B, Hc, Hc, scale);
        int Mc = B * Hc * Hc;
        gemm(T1h, 256, outWh[i] + 256, 768, out_b[i] + 256, T2, nullptr, 256,
             Mc, 256, 256, 0);
        int n = Mrows[i] * 256;
        up_add_kernel<<<(n + 255) / 256, 256, 0, stream>>>(T2, ACC[i], B, Hc, Hc);
    }

    // 5) bottom-up: coarse queries vs stride-2 fine K/V
    for (int i = 0; i < 2; ++i) {
        int j = i + 1;
        int Hj = Hd[j], Hi = Hd[i];
        int qoff = (j == 2) ? 1280 : 1536;
        int posrows = (j == 1) ? 768 : 512;
        int tot = B * Hj * Hj * 8;
        local_attn_kernel<<<(tot + 255) / 256, 256, 0, stream>>>(
            P[j], qoff, Pw[j], Hj, Hj, 1,
            P[i], Pw[i] - 512, Pw[i] - 256, Pw[i], Hi, Hi, 2,
            pos[j] + (size_t)(posrows - 256) * 9, T1h, B, Hj, Hj, scale);
        gemm(T1h, 256, outWh[j] + 512, 768, out_b[j] + 512, ACC[j], nullptr, 256,
             Mrows[j], 256, 256, GEMM_BETA);
    }

    // 6) residual + LN1 + FFN + residual + LN2 (projections dead; reuse as scratch)
    for (int i = 0; i < 3; ++i) {
        int M = Mrows[i];
        ln_kernel<<<(M + 7) / 8, 256, 0, stream>>>(ACC[i], nullptr, ln1_w[i], ln1_b[i],
                                                   XBUF, XBh, M);
        gemm(XBh, 256, f1Wh[i], 256, f1_b[i], nullptr, HBh, 1024,
             M, 1024, 256, GEMM_RELU);
        gemm(HBh, 1024, f2Wh[i], 1024, f2_b[i], T2, nullptr, 256,
             M, 256, 1024, 0);
        ln_kernel<<<(M + 7) / 8, 256, 0, stream>>>(XBUF, T2, ln2_w[i], ln2_b[i],
                                                   outp + outoff[i], nullptr, M);
    }
}
